// RealMambaCore_1073741824059
// MI455X (gfx1250) — matrix operand order repaired and checked by reading, not compiled, not run on hardware
//
#include <hip/hip_runtime.h>
#include <hip/hip_bf16.h>

#define HIDDEN 1024
#define SEQ    2048
#define NBATCH 4
#define MROWS  (NBATCH * SEQ)   // 8192 rows

typedef __attribute__((ext_vector_type(16))) __bf16 v16bf;
typedef __attribute__((ext_vector_type(8)))  float  v8f;

// ---------------------------------------------------------------- helpers
__device__ __forceinline__ __bf16 f2bf(float f) {
    union { float f; unsigned u; } a; a.f = f;
    unsigned r = a.u + 0x7FFFu + ((a.u >> 16) & 1u);   // RNE
    union { unsigned short s; __bf16 b; } o; o.s = (unsigned short)(r >> 16);
    return o.b;
}

__device__ __forceinline__ float silu_f(float x) {
    return x / (1.0f + __expf(-x));
}

// A fragment, 16x32 bf16 (ISA 7.12.2): lanes 0-15 -> M=0..15, lanes 16-31 same
// rows; VGPR i (elem pair 2i,2i+1): K = (i>=4?16:0) + (lane>=16?8:0) + 2*(i&3).
__device__ __forceinline__ v16bf load_a_frag(const __bf16* __restrict__ s,
                                             int ld, int row0, int ks, int lane) {
    v16bf a;
    const int r  = row0 + (lane & 15);
    const int kb = ks * 32 + ((lane >> 4) << 3);
    const __bf16* p = s + r * ld + kb;
#pragma unroll
    for (int i = 0; i < 4; ++i) {
        a[2*i]     = p[2*i];
        a[2*i + 1] = p[2*i + 1];
    }
#pragma unroll
    for (int i = 0; i < 4; ++i) {
        a[8 + 2*i] = p[16 + 2*i];
        a[9 + 2*i] = p[16 + 2*i + 1];
    }
    return a;
}

// B fragment, 32x16 bf16: lane l holds N=l%16; element e has
// K = (e>=8?16:0) + (lane>=16?8:0) + (e&7), matching the A fragment.
__device__ __forceinline__ v16bf load_b_frag(const __bf16* __restrict__ s,
                                             int ld, int col0, int ks, int lane) {
    v16bf b;
    const __bf16* p = s + (ks * 32 + ((lane >> 4) << 3)) * ld + col0 + (lane & 15);
#pragma unroll
    for (int e = 0; e < 16; ++e) b[e] = p[((e & 7) + ((e >> 3) << 4)) * ld];
    return b;
}

__device__ __forceinline__ v8f zero_v8f() {
    v8f z;
#pragma unroll
    for (int e = 0; e < 8; ++e) z[e] = 0.0f;
    return z;
}

// ---------------------------------------------------------------- fp32 -> bf16
__global__ __launch_bounds__(256) void cvt_kernel(const float* __restrict__ s,
                                                  __bf16* __restrict__ d, int n) {
    int i = (blockIdx.x * 256 + threadIdx.x) * 4;
    if (i < n) {
        float4 f = *(const float4*)(s + i);
        d[i + 0] = f2bf(f.x);
        d[i + 1] = f2bf(f.y);
        d[i + 2] = f2bf(f.z);
        d[i + 3] = f2bf(f.w);
    }
}

// ---------------------------------------------------------------- dual GEMM
// out = silu(X@Wa + biasA) * (X@Wb + biasB)   (shared X tile for both GEMMs)
// BM=128, BN=64, BK=64; 8 waves 4x2; wave tile 32x32 per output.
template <bool IN_F32, bool OUT_F32>
__global__ __launch_bounds__(256) void dual_gemm_kernel(
    const void* __restrict__ Xv,
    const __bf16* __restrict__ Wa, const __bf16* __restrict__ Wb,
    const float* __restrict__ biasA, const float* __restrict__ biasB,
    void* __restrict__ Ov)
{
    constexpr int BM = 128, BN = 64, BK = 64, LDX = BK + 4, LDW = BN + 4;
    __shared__ __align__(16) __bf16 sX[BM][LDX];
    __shared__ __align__(16) __bf16 sWa[BK][LDW];
    __shared__ __align__(16) __bf16 sWb[BK][LDW];

    const int tid  = threadIdx.x, lane = tid & 31, wave = tid >> 5;
    const int wm = wave & 3, wn = wave >> 2;
    const int rowBase = wm * 32, colBase = wn * 32;
    const int blockRow = blockIdx.x * BM, blockCol = blockIdx.y * BN;

    v8f accA[2][2], accB[2][2];
#pragma unroll
    for (int m = 0; m < 2; ++m)
#pragma unroll
        for (int n = 0; n < 2; ++n) { accA[m][n] = zero_v8f(); accB[m][n] = zero_v8f(); }

    for (int kt = 0; kt < HIDDEN / BK; ++kt) {
        // X tile 128x64 (8 vec4 per thread)
#pragma unroll
        for (int i = 0; i < 8; ++i) {
            int vi = tid + i * 256;
            int r = vi >> 4, c4 = (vi & 15) << 2;
            if constexpr (IN_F32) {
                float4 f = *(const float4*)((const float*)Xv +
                            (size_t)(blockRow + r) * HIDDEN + kt * BK + c4);
                __bf16* d = &sX[r][c4];
                d[0] = f2bf(f.x); d[1] = f2bf(f.y); d[2] = f2bf(f.z); d[3] = f2bf(f.w);
            } else {
                *(uint2*)&sX[r][c4] = *(const uint2*)((const __bf16*)Xv +
                            (size_t)(blockRow + r) * HIDDEN + kt * BK + c4);
            }
        }
        // two weight tiles 64x64 (4 vec4 per thread each)
#pragma unroll
        for (int i = 0; i < 4; ++i) {
            int vi = tid + i * 256;
            int r = vi >> 4, c4 = (vi & 15) << 2;
            size_t g = (size_t)(kt * BK + r) * HIDDEN + blockCol + c4;
            *(uint2*)&sWa[r][c4] = *(const uint2*)(Wa + g);
            *(uint2*)&sWb[r][c4] = *(const uint2*)(Wb + g);
        }
        __syncthreads();
#pragma unroll
        for (int ks = 0; ks < 2; ++ks) {
            v16bf af[2];
#pragma unroll
            for (int m = 0; m < 2; ++m)
                af[m] = load_a_frag(&sX[0][0], LDX, rowBase + m * 16, ks, lane);
#pragma unroll
            for (int n = 0; n < 2; ++n) {
                v16bf bfa = load_b_frag(&sWa[0][0], LDW, colBase + n * 16, ks, lane);
                v16bf bfb = load_b_frag(&sWb[0][0], LDW, colBase + n * 16, ks, lane);
#pragma unroll
                for (int m = 0; m < 2; ++m) {
                    accA[m][n] = __builtin_amdgcn_wmma_f32_16x16x32_bf16(
                        false, af[m], false, bfa, (short)0, accA[m][n], false, false);
                    accB[m][n] = __builtin_amdgcn_wmma_f32_16x16x32_bf16(
                        false, af[m], false, bfb, (short)0, accB[m][n], false, false);
                }
            }
        }
        __syncthreads();
    }
    // epilogue: silu(a+ba) * (b+bb)
    const int mOff = (lane >> 4) << 3, nOff = lane & 15;
#pragma unroll
    for (int m = 0; m < 2; ++m)
#pragma unroll
        for (int n = 0; n < 2; ++n) {
            int gc = blockCol + colBase + n * 16 + nOff;
            float ba = biasA[gc], bb = biasB[gc];
#pragma unroll
            for (int j = 0; j < 8; ++j) {
                int gr = blockRow + rowBase + m * 16 + mOff + j;
                float val = silu_f(accA[m][n][j] + ba) * (accB[m][n][j] + bb);
                if constexpr (OUT_F32)
                    ((float*)Ov)[(size_t)gr * HIDDEN + gc] = val;
                else
                    ((__bf16*)Ov)[(size_t)gr * HIDDEN + gc] = f2bf(val);
            }
        }
}

// ---------------------------------------------------------------- single GEMM
// out = X@W + bias.  BM=128, BN=128, BK=64; 8 waves 4x2; wave tile 32x64.
template <bool OUT_F32>
__global__ __launch_bounds__(256) void gemm128_kernel(
    const __bf16* __restrict__ X, const __bf16* __restrict__ W,
    const float* __restrict__ bias, void* __restrict__ Ov)
{
    constexpr int BM = 128, BN = 128, BK = 64, LDX = BK + 4, LDW = BN + 4;
    __shared__ __align__(16) __bf16 sX[BM][LDX];
    __shared__ __align__(16) __bf16 sW[BK][LDW];

    const int tid = threadIdx.x, lane = tid & 31, wave = tid >> 5;
    const int wm = wave & 3, wn = wave >> 2;
    const int rowBase = wm * 32, colBase = wn * 64;
    const int blockRow = blockIdx.x * BM, blockCol = blockIdx.y * BN;

    v8f acc[2][4];
#pragma unroll
    for (int m = 0; m < 2; ++m)
#pragma unroll
        for (int n = 0; n < 4; ++n) acc[m][n] = zero_v8f();

    for (int kt = 0; kt < HIDDEN / BK; ++kt) {
#pragma unroll
        for (int i = 0; i < 8; ++i) {           // X tile 128x64
            int vi = tid + i * 256;
            int r = vi >> 4, c4 = (vi & 15) << 2;
            *(uint2*)&sX[r][c4] = *(const uint2*)(X +
                (size_t)(blockRow + r) * HIDDEN + kt * BK + c4);
        }
#pragma unroll
        for (int i = 0; i < 8; ++i) {           // W tile 64x128
            int vi = tid + i * 256;
            int r = vi >> 5, c4 = (vi & 31) << 2;
            *(uint2*)&sW[r][c4] = *(const uint2*)(W +
                (size_t)(kt * BK + r) * HIDDEN + blockCol + c4);
        }
        __syncthreads();
#pragma unroll
        for (int ks = 0; ks < 2; ++ks) {
            v16bf af[2];
#pragma unroll
            for (int m = 0; m < 2; ++m)
                af[m] = load_a_frag(&sX[0][0], LDX, rowBase + m * 16, ks, lane);
#pragma unroll
            for (int n = 0; n < 4; ++n) {
                v16bf bfr = load_b_frag(&sW[0][0], LDW, colBase + n * 16, ks, lane);
#pragma unroll
                for (int m = 0; m < 2; ++m)
                    acc[m][n] = __builtin_amdgcn_wmma_f32_16x16x32_bf16(
                        false, af[m], false, bfr, (short)0, acc[m][n], false, false);
            }
        }
        __syncthreads();
    }
    const int mOff = (lane >> 4) << 3, nOff = lane & 15;
#pragma unroll
    for (int m = 0; m < 2; ++m)
#pragma unroll
        for (int n = 0; n < 4; ++n) {
            int gc = blockCol + colBase + n * 16 + nOff;
            float bs = bias[gc];
#pragma unroll
            for (int j = 0; j < 8; ++j) {
                int gr = blockRow + rowBase + m * 16 + mOff + j;
                float val = acc[m][n][j] + bs;
                if constexpr (OUT_F32)
                    ((float*)Ov)[(size_t)gr * HIDDEN + gc] = val;
                else
                    ((__bf16*)Ov)[(size_t)gr * HIDDEN + gc] = f2bf(val);
            }
        }
}

// ---------------------------------------------------------------- SSM scan
// FFT conv with K[t]=Bp*A^t  ==  y[t] = A*y[t-1] + Bp*u[t]; v = Cp*y + Dp*u.
__global__ __launch_bounds__(256) void scan_kernel(
    const float* __restrict__ u, const float* __restrict__ A,
    const float* __restrict__ Bp, const float* __restrict__ Cp,
    const float* __restrict__ Dp, float* __restrict__ v)
{
    int idx = blockIdx.x * 256 + threadIdx.x;      // 0 .. NBATCH*HIDDEN-1
    int b = idx >> 10;                             // / HIDDEN
    int d = idx & (HIDDEN - 1);
    float a = A[d], bp = Bp[d], cp = Cp[d], dp = Dp[d];
    const float* up = u + (size_t)b * SEQ * HIDDEN + d;
    float*       vp = v + (size_t)b * SEQ * HIDDEN + d;
    float y = 0.0f;
    for (int t = 0; t < SEQ; ++t) {
        float uu = up[(size_t)t * HIDDEN];
        y = a * y + bp * uu;
        vp[(size_t)t * HIDDEN] = cp * y + dp * uu;
    }
}

// ---------------------------------------------------------------- LayerNorm
__global__ __launch_bounds__(256) void ln_kernel(
    const float* __restrict__ v, const float* __restrict__ gamma,
    const float* __restrict__ beta, __bf16* __restrict__ out)
{
    __shared__ float rs[256], rq[256];
    int row = blockIdx.x, t = threadIdx.x;
    const float* p = v + (size_t)row * HIDDEN + t * 4;
    float4 x = *(const float4*)p;
    rs[t] = x.x + x.y + x.z + x.w;
    rq[t] = x.x * x.x + x.y * x.y + x.z * x.z + x.w * x.w;
    __syncthreads();
    for (int off = 128; off > 0; off >>= 1) {
        if (t < off) { rs[t] += rs[t + off]; rq[t] += rq[t + off]; }
        __syncthreads();
    }
    float mu   = rs[0] * (1.0f / HIDDEN);
    float var  = rq[0] * (1.0f / HIDDEN) - mu * mu;
    float rstd = rsqrtf(var + 1e-3f);
    int d0 = t * 4;
    __bf16* o = out + (size_t)row * HIDDEN + d0;
    o[0] = f2bf((x.x - mu) * rstd * gamma[d0 + 0] + beta[d0 + 0]);
    o[1] = f2bf((x.y - mu) * rstd * gamma[d0 + 1] + beta[d0 + 1]);
    o[2] = f2bf((x.z - mu) * rstd * gamma[d0 + 2] + beta[d0 + 2]);
    o[3] = f2bf((x.w - mu) * rstd * gamma[d0 + 3] + beta[d0 + 3]);
}

// ---------------------------------------------------------------- launch
extern "C" void kernel_launch(void* const* d_in, const int* in_sizes, int n_in,
                              void* d_out, int out_size, void* d_ws, size_t ws_size,
                              hipStream_t stream)
{
    const float* x     = (const float*)d_in[0];
    const float* Wg    = (const float*)d_in[1];
    const float* bg    = (const float*)d_in[2];
    const float* Wi    = (const float*)d_in[3];
    const float* bi    = (const float*)d_in[4];
    const float* A     = (const float*)d_in[5];
    const float* Bp    = (const float*)d_in[6];
    const float* Cp    = (const float*)d_in[7];
    const float* Dp    = (const float*)d_in[8];
    const float* gamma = (const float*)d_in[9];
    const float* beta  = (const float*)d_in[10];
    const float* Wfg   = (const float*)d_in[11];
    const float* bfg   = (const float*)d_in[12];
    const float* Wfu   = (const float*)d_in[13];
    const float* bfu   = (const float*)d_in[14];
    const float* Wfd   = (const float*)d_in[15];
    const float* bfd   = (const float*)d_in[16];
    const float* Wo    = (const float*)d_in[17];
    const float* bo    = (const float*)d_in[18];

    // workspace layout: 6 bf16 weights (12MB) + u,v fp32 (64MB) + 3 bf16 acts (48MB)
    char* ws = (char*)d_ws;
    const size_t WB = (size_t)HIDDEN * HIDDEN * sizeof(__bf16);
    __bf16* wg_b  = (__bf16*)(ws + 0 * WB);
    __bf16* wi_b  = (__bf16*)(ws + 1 * WB);
    __bf16* wfg_b = (__bf16*)(ws + 2 * WB);
    __bf16* wfu_b = (__bf16*)(ws + 3 * WB);
    __bf16* wfd_b = (__bf16*)(ws + 4 * WB);
    __bf16* wo_b  = (__bf16*)(ws + 5 * WB);
    size_t off = 6 * WB;
    float*  u   = (float*)(ws + off);  off += (size_t)MROWS * HIDDEN * sizeof(float);
    float*  v   = (float*)(ws + off);  off += (size_t)MROWS * HIDDEN * sizeof(float);
    __bf16* yln = (__bf16*)(ws + off); off += (size_t)MROWS * HIDDEN * sizeof(__bf16);
    __bf16* h   = (__bf16*)(ws + off); off += (size_t)MROWS * HIDDEN * sizeof(__bf16);
    __bf16* z   = (__bf16*)(ws + off);

    const int nW = HIDDEN * HIDDEN;
    dim3 cb(256), cg(nW / 4 / 256);
    cvt_kernel<<<cg, cb, 0, stream>>>(Wg,  wg_b,  nW);
    cvt_kernel<<<cg, cb, 0, stream>>>(Wi,  wi_b,  nW);
    cvt_kernel<<<cg, cb, 0, stream>>>(Wfg, wfg_b, nW);
    cvt_kernel<<<cg, cb, 0, stream>>>(Wfu, wfu_b, nW);
    cvt_kernel<<<cg, cb, 0, stream>>>(Wfd, wfd_b, nW);
    cvt_kernel<<<cg, cb, 0, stream>>>(Wo,  wo_b,  nW);

    dim3 blk(256);
    // stage 1: u = silu(x@Wg+bg) * (x@Wi+bi)
    dual_gemm_kernel<true, true><<<dim3(MROWS / 128, HIDDEN / 64), blk, 0, stream>>>(
        (const void*)x, wg_b, wi_b, bg, bi, (void*)u);
    // stage 2: linear-recurrence scan (== the FFT causal conv) + C*y + D*u
    scan_kernel<<<dim3((NBATCH * HIDDEN) / 256), blk, 0, stream>>>(u, A, Bp, Cp, Dp, v);
    // stage 3: LayerNorm -> bf16
    ln_kernel<<<dim3(MROWS), blk, 0, stream>>>(v, gamma, beta, yln);
    // stage 4: h = silu(y@Wfg+bfg) * (y@Wfu+bfu)
    dual_gemm_kernel<false, false><<<dim3(MROWS / 128, HIDDEN / 64), blk, 0, stream>>>(
        (const void*)yln, wfg_b, wfu_b, bfg, bfu, (void*)h);
    // stage 5: z = h@Wfd + bfd
    gemm128_kernel<false><<<dim3(MROWS / 128, HIDDEN / 128), blk, 0, stream>>>(
        h, wfd_b, bfd, (void*)z);
    // stage 6: out = z@Wo + bo  (fp32)
    gemm128_kernel<true><<<dim3(MROWS / 128, HIDDEN / 128), blk, 0, stream>>>(
        z, wo_b, bo, d_out);
}
